// GraphConvolution_1219770712466
// MI455X (gfx1250) — matrix operand order repaired and checked by reading, not compiled, not run on hardware
//
#include <hip/hip_runtime.h>
#include <cstddef>

// ---------------------------------------------------------------------------
// GraphConvolution (node_layer) for MI455X / gfx1250, wave32 + WMMA f16.
//
//   d[b,e]      = H_e[b,e,:] . p
//   mult[b,v,w] = sum_e (T[b,v,e]*d[b,e]) * T[b,w,e]      <-- 275 GFLOP, WMMA
//   adjA        = (diag->1)(mult) * adj_v                  (stored f16)
//   out         = adjA @ (H_v @ weight) + bias             <-- WMMA
//   outputs     = [out (B,NV,128) f32 , H_e copy]
//
// Workspace layout (needs ~71.6 MB):
//   [0,            131072)                     d      f32  [B,NE]
//   [131072,       131072+4  MB)               HWt16  f16  [B,128,NV] (transposed)
//   [131072+4 MB,  131072+4 MB+64 MB)          adjA16 f16  [B,NV,NV]
// ---------------------------------------------------------------------------

typedef __attribute__((ext_vector_type(16))) _Float16 v16h;
typedef __attribute__((ext_vector_type(8)))  _Float16 v8h;
typedef __attribute__((ext_vector_type(4)))  _Float16 v4h;
typedef __attribute__((ext_vector_type(8)))  float    v8f;

#define B_    8
#define NV_   2048
#define NE_   4096
#define INV_  128
#define OUTV_ 128
#define INE_  64

__device__ __forceinline__ v8f vzero8() {
  v8f z;
#pragma unroll
  for (int i = 0; i < 8; ++i) z[i] = 0.0f;
  return z;
}

__device__ __forceinline__ v16h vcat(v8h a, v8h b) {
  v16h r;
#pragma unroll
  for (int i = 0; i < 8; ++i) { r[i] = a[i]; r[i + 8] = b[i]; }
  return r;
}

// ---------------------------------------------------------------- kernel 1 --
__global__ void k_dot(const float* __restrict__ He, const float* __restrict__ p,
                      float* __restrict__ d) {
  int idx = blockIdx.x * blockDim.x + threadIdx.x;        // (b,e) flat
  if (idx >= B_ * NE_) return;
  const float4* row = (const float4*)(He + (size_t)idx * INE_);
  const float4* p4  = (const float4*)p;
  float s = 0.0f;
#pragma unroll
  for (int i = 0; i < INE_ / 4; ++i) {
    float4 a = row[i], b = p4[i];
    s += a.x * b.x + a.y * b.y + a.z * b.z + a.w * b.w;
  }
  d[idx] = s;
}

// ---------------------------------------------------------------- kernel 2 --
// HWt16[b][o][v] = f16( sum_k H_v[b,v,k] * W[k,o] )   (transposed store)
__global__ void k_hw(const float* __restrict__ Hv, const float* __restrict__ W,
                     _Float16* __restrict__ HWt) {
  int bv = blockIdx.x;                     // one block per (b,v) row
  int b = bv / NV_, v = bv % NV_;
  __shared__ float hv[INV_];
  int o = threadIdx.x;                     // 128 threads
  hv[o] = Hv[(size_t)bv * INV_ + o];
  __syncthreads();
  float s = 0.0f;
#pragma unroll 8
  for (int k = 0; k < INV_; ++k) s += hv[k] * W[k * OUTV_ + o];
  HWt[((size_t)b * OUTV_ + o) * NV_ + v] = (_Float16)s;
}

// ---------------------------------------------------------------- kernel 3 --
// adjA16[b,v,w] = f16( blend_diag(sum_e (T[v,e]*d[e])*T[w,e]) * adj_v[v,w] )
#define BM3 128
#define BN3 256
#define BK3 32
#define LDA3 40   // halves; 80 B row stride: 16B-aligned, bank-spread

__global__ __launch_bounds__(256, 1)
void k_mult(const float* __restrict__ T, const float* __restrict__ adj_v,
            const float* __restrict__ d, _Float16* __restrict__ adjA) {
  const int b      = blockIdx.z;
  const int vblock = blockIdx.y * BM3;
  const int wblock = blockIdx.x * BN3;
  const float* Tb  = T + (size_t)b * NV_ * NE_;

  __shared__ _Float16 Asub[BM3][LDA3];
  __shared__ _Float16 Bsub[BN3][LDA3];
  __shared__ float    dsh[NE_];

  const int tid  = threadIdx.x;
  const int lane = tid & 31;
  const int wave = tid >> 5;
  const int wrow = wave >> 2;        // 0..1  -> 64-row slab
  const int wcol = wave & 3;         // 0..3  -> 64-col slab
  const int l    = lane & 15;
  const int hi   = lane >> 4;        // half-wave selector

  // stage d[b,:] once (16 KB)
  {
    const float4* src = (const float4*)(d + (size_t)b * NE_);
    float4* dst = (float4*)dsh;
#pragma unroll
    for (int i = 0; i < NE_ / 4 / 256; ++i) dst[tid + i * 256] = src[tid + i * 256];
  }

  v8f acc[4][4];
#pragma unroll
  for (int mt = 0; mt < 4; ++mt)
#pragma unroll
    for (int nt = 0; nt < 4; ++nt) acc[mt][nt] = vzero8();

  const int arow = tid >> 1;           // 0..127, 2 threads/row
  const int acol = (tid & 1) * 16;     // 16 floats each
  const int brow = tid;                // 0..255, full row each
  const float* aBase = Tb + (size_t)(vblock + arow) * NE_ + acol;
  const float* bBase = Tb + (size_t)(wblock + brow) * NE_;

  // ---- prime staging registers with chunk 0 ----
  float4 aReg[4], bReg[8];
#pragma unroll
  for (int i = 0; i < 4; ++i) aReg[i] = ((const float4*)aBase)[i];
#pragma unroll
  for (int i = 0; i < 8; ++i) bReg[i] = ((const float4*)bBase)[i];

  __syncthreads();                     // dsh visible

  for (int kk = 0; kk < NE_; kk += BK3) {
    // ---- f32 -> f16 LDS staging of the chunk held in registers ----
#pragma unroll
    for (int i = 0; i < 4; ++i) {
      float4 a  = aReg[i];
      int    c  = acol + i * 4;
      float4 dv = *(const float4*)(dsh + kk + c);
      v4h h;
      h[0] = (_Float16)(a.x * dv.x); h[1] = (_Float16)(a.y * dv.y);
      h[2] = (_Float16)(a.z * dv.z); h[3] = (_Float16)(a.w * dv.w);
      *(v4h*)(&Asub[arow][c]) = h;
    }
#pragma unroll
    for (int i = 0; i < 8; ++i) {
      float4 v = bReg[i];
      v4h h;
      h[0] = (_Float16)v.x; h[1] = (_Float16)v.y;
      h[2] = (_Float16)v.z; h[3] = (_Float16)v.w;
      *(v4h*)(&Bsub[brow][i * 4]) = h;
    }
    __syncthreads();

    // ---- prefetch next chunk; loads fly while the WMMAs below execute ----
    if (kk + BK3 < NE_) {
      const float4* an = (const float4*)(aBase + kk + BK3);
      const float4* bn = (const float4*)(bBase + kk + BK3);
#pragma unroll
      for (int i = 0; i < 4; ++i) aReg[i] = an[i];
#pragma unroll
      for (int i = 0; i < 8; ++i) bReg[i] = bn[i];
    }

    // ---- fragments per ISA layouts ----
    // A 16x32 f16: lanes0-15 K{0..7,16..23}, lanes16-31 K{8..15,24..31}
    const int ak0 = hi ? 8 : 0;
    v16h afrag[4], bfrag[4];
#pragma unroll
    for (int mt = 0; mt < 4; ++mt) {
      int m = wrow * 64 + mt * 16 + l;
      afrag[mt] = vcat(*(const v8h*)(&Asub[m][ak0]),
                       *(const v8h*)(&Asub[m][ak0 + 16]));
    }
    // B 32x16 f16: lanes0-15 K{0..7,16..23}, lanes16-31 K{8..15,24..31}
    const int bk0 = hi ? 8 : 0;
#pragma unroll
    for (int nt = 0; nt < 4; ++nt) {
      int n = wcol * 64 + nt * 16 + l;
      bfrag[nt] = vcat(*(const v8h*)(&Bsub[n][bk0]),
                       *(const v8h*)(&Bsub[n][bk0 + 16]));
    }
#pragma unroll
    for (int mt = 0; mt < 4; ++mt)
#pragma unroll
      for (int nt = 0; nt < 4; ++nt)
        acc[mt][nt] = __builtin_amdgcn_wmma_f32_16x16x32_f16(
            false, afrag[mt], false, bfrag[nt], (short)0, acc[mt][nt], false, false);
    __syncthreads();
  }

  // ---- epilogue: diag->1, *adj_v, f16 store (adj_v loads batched) ----
  const float*   av = adj_v + (size_t)b * NV_ * NV_;
  _Float16*      ob = adjA  + (size_t)b * NV_ * NV_;
#pragma unroll
  for (int mt = 0; mt < 4; ++mt) {
    const int gmb = vblock + wrow * 64 + mt * 16 + (hi ? 8 : 0);
    float avv[4][8];
#pragma unroll
    for (int nt = 0; nt < 4; ++nt) {
      int gn = wblock + wcol * 64 + nt * 16 + l;
#pragma unroll
      for (int r = 0; r < 8; ++r)
        avv[nt][r] = av[(size_t)(gmb + r) * NV_ + gn];   // 32 loads in flight
    }
#pragma unroll
    for (int nt = 0; nt < 4; ++nt) {
      int gn = wblock + wcol * 64 + nt * 16 + l;
#pragma unroll
      for (int r = 0; r < 8; ++r) {
        int gm  = gmb + r;
        float v = acc[mt][nt][r];
        if (gm == gn) v = 1.0f;
        ob[(size_t)gm * NV_ + gn] = (_Float16)(v * avv[nt][r]);
      }
    }
  }
}

// ---------------------------------------------------------------- kernel 4 --
// out[b,v,o] = sum_w adjA16[b,v,w]*HWt16[b,o,w] + bias[o]
#define BM4 128
#define BK4 32
#define LDA4 40

__global__ __launch_bounds__(256, 1)
void k_gemm2(const _Float16* __restrict__ adjA, const _Float16* __restrict__ HWt,
             const float* __restrict__ bias, float* __restrict__ out) {
  const int b      = blockIdx.z;
  const int vblock = blockIdx.y * BM4;

  __shared__ _Float16 Asub[BM4][LDA4];
  __shared__ _Float16 Bsub[OUTV_][LDA4];

  const int tid  = threadIdx.x;
  const int lane = tid & 31;
  const int wave = tid >> 5;
  const int wrow = wave >> 2;        // 0..1 -> 64 rows
  const int wcol = wave & 3;         // 0..3 -> 32 cols
  const int l    = lane & 15;
  const int hi   = lane >> 4;

  v8f acc[4][2];
#pragma unroll
  for (int mt = 0; mt < 4; ++mt)
#pragma unroll
    for (int nt = 0; nt < 2; ++nt) acc[mt][nt] = vzero8();

  const _Float16* Ab = adjA + (size_t)b * NV_ * NV_;
  const _Float16* Bb = HWt  + (size_t)b * OUTV_ * NV_;
  const int row = tid >> 1;            // 0..127
  const int seg = (tid & 1) * 16;      // 16 halves each
  const _Float16* aBase = Ab + (size_t)(vblock + row) * NV_ + seg;
  const _Float16* bBase = Bb + (size_t)row * NV_ + seg;

  v8h aReg[2], bReg[2];
  aReg[0] = ((const v8h*)aBase)[0]; aReg[1] = ((const v8h*)aBase)[1];
  bReg[0] = ((const v8h*)bBase)[0]; bReg[1] = ((const v8h*)bBase)[1];

  for (int kk = 0; kk < NV_; kk += BK4) {
    *(v8h*)(&Asub[row][seg])     = aReg[0];
    *(v8h*)(&Asub[row][seg + 8]) = aReg[1];
    *(v8h*)(&Bsub[row][seg])     = bReg[0];
    *(v8h*)(&Bsub[row][seg + 8]) = bReg[1];
    __syncthreads();

    if (kk + BK4 < NV_) {
      const v8h* an = (const v8h*)(aBase + kk + BK4);
      const v8h* bn = (const v8h*)(bBase + kk + BK4);
      aReg[0] = an[0]; aReg[1] = an[1];
      bReg[0] = bn[0]; bReg[1] = bn[1];
    }

    const int ak0 = hi ? 8 : 0;
    const int bk0 = hi ? 8 : 0;
    v16h afrag[4], bfrag[2];
#pragma unroll
    for (int mt = 0; mt < 4; ++mt) {
      int m = wrow * 64 + mt * 16 + l;
      afrag[mt] = vcat(*(const v8h*)(&Asub[m][ak0]),
                       *(const v8h*)(&Asub[m][ak0 + 16]));
    }
#pragma unroll
    for (int nt = 0; nt < 2; ++nt) {
      int n = wcol * 32 + nt * 16 + l;
      bfrag[nt] = vcat(*(const v8h*)(&Bsub[n][bk0]),
                       *(const v8h*)(&Bsub[n][bk0 + 16]));
    }
#pragma unroll
    for (int mt = 0; mt < 4; ++mt)
#pragma unroll
      for (int nt = 0; nt < 2; ++nt)
        acc[mt][nt] = __builtin_amdgcn_wmma_f32_16x16x32_f16(
            false, afrag[mt], false, bfrag[nt], (short)0, acc[mt][nt], false, false);
    __syncthreads();
  }

  float* ob = out + (size_t)b * NV_ * OUTV_;
#pragma unroll
  for (int mt = 0; mt < 4; ++mt)
#pragma unroll
    for (int nt = 0; nt < 2; ++nt) {
      int gn = wcol * 32 + nt * 16 + l;
      float bs = bias[gn];
#pragma unroll
      for (int r = 0; r < 8; ++r) {
        int gm = vblock + wrow * 64 + mt * 16 + r + (hi ? 8 : 0);
        ob[(size_t)gm * OUTV_ + gn] = acc[mt][nt][r] + bs;
      }
    }
}

// --------------------------------------------------------------- dispatch --
extern "C" void kernel_launch(void* const* d_in, const int* in_sizes, int n_in,
                              void* d_out, int out_size, void* d_ws, size_t ws_size,
                              hipStream_t stream) {
  (void)in_sizes; (void)n_in; (void)out_size; (void)ws_size;
  const float* Hv    = (const float*)d_in[0];
  const float* He    = (const float*)d_in[1];
  // d_in[2] = adj_e: unused by the reference result
  const float* adj_v = (const float*)d_in[3];
  const float* T     = (const float*)d_in[4];
  const float* W     = (const float*)d_in[5];
  const float* p     = (const float*)d_in[6];
  const float* bias  = (const float*)d_in[7];
  float* out = (float*)d_out;

  char* ws = (char*)d_ws;
  float*     d_d  = (float*)ws;                                   // 131072 B
  _Float16*  HWt  = (_Float16*)(ws + 131072);                     // 4 MB
  _Float16*  adjA = (_Float16*)(ws + 131072 + (size_t)4194304);   // 64 MB

  // Output #2: H_e passthrough (independent; overlap with compute)
  hipMemcpyAsync(out + (size_t)B_ * NV_ * OUTV_, He,
                 (size_t)B_ * NE_ * INE_ * sizeof(float),
                 hipMemcpyDeviceToDevice, stream);

  k_dot<<<(B_ * NE_) / 256, 256, 0, stream>>>(He, p, d_d);
  k_hw<<<B_ * NV_, INV_, 0, stream>>>(Hv, W, HWt);
  k_mult<<<dim3(NV_ / BN3, NV_ / BM3, B_), 256, 0, stream>>>(T, adj_v, d_d, adjA);
  k_gemm2<<<dim3(1, NV_ / BM4, B_), 256, 0, stream>>>(adjA, HWt, bias, out);
}
